// HiCL_35433480192893
// MI455X (gfx1250) — matrix operand order repaired and checked by reading, not compiled, not run on hardware
//
#include <hip/hip_runtime.h>
#include <hip/hip_bf16.h>

typedef __bf16 bf16_t;
typedef __attribute__((ext_vector_type(16))) __bf16 v16bf;
typedef __attribute__((ext_vector_type(4)))  __bf16 v4bf;
typedef __attribute__((ext_vector_type(8)))  float  v8f;
typedef __attribute__((ext_vector_type(4)))  float  v4f;

union FragBF { v16bf v; v4f f[2]; };

#define N_ITEMS     8192
#define FDIM        256
#define M_ITEMS     20000
#define TRACE_DEPTH 4
#define NUM_CLASSES 1000
#define INV_T       5.0f
#define LOSS_SCALE  0.001f

// ---------------------------------------------------------------------------
// fp32 -> bf16 conversion; A is pre-scaled by 1/T so WMMA output == logits
// ---------------------------------------------------------------------------
__global__ __launch_bounds__(256)
void cvt_bf16_4(const float* __restrict__ src, bf16_t* __restrict__ dst,
                int n, float scale) {
  int i = (blockIdx.x * blockDim.x + threadIdx.x) * 4;
  if (i < n) {
    v4f v = *(const v4f*)(src + i);
    v4bf o;
    o[0] = (bf16_t)(v[0] * scale);
    o[1] = (bf16_t)(v[1] * scale);
    o[2] = (bf16_t)(v[2] * scale);
    o[3] = (bf16_t)(v[3] * scale);
    *(v4bf*)(dst + i) = o;
  }
}

// ---------------------------------------------------------------------------
// Fused bf16 WMMA GEMM + online logsumexp, software-pipelined B tiles.
// ---------------------------------------------------------------------------
__device__ __forceinline__ void load_btile(FragBF b[8], const bf16_t* bp) {
#pragma unroll
  for (int ks = 0; ks < 8; ++ks) {
    b[ks].f[0] = *(const v4f*)(bp + ks * 32);      // K = 32ks+8h   .. +7
    b[ks].f[1] = *(const v4f*)(bp + ks * 32 + 16); // K = 32ks+16+8h.. +7
  }
}

__device__ __forceinline__ void compute_tile(const FragBF a[8],
                                             const FragBF b[8],
                                             float rmax[8], float rsum[8]) {
  v8f c = {};
#pragma unroll
  for (int ks = 0; ks < 8; ++ks)
    c = __builtin_amdgcn_wmma_f32_16x16x32_bf16(
            /*neg_a=*/false, a[ks].v, /*neg_b=*/false, b[ks].v,
            /*c_mod=*/(short)0, c, /*reuse_a=*/false, /*reuse_b=*/false);
  // online logsumexp update (TRANS v_exp co-executes with next tile's WMMAs)
#pragma unroll
  for (int j = 0; j < 8; ++j) {
    float v = c[j];
    float m = fmaxf(rmax[j], v);
    rsum[j] = rsum[j] * __expf(rmax[j] - m) + __expf(v - m);
    rmax[j] = m;
  }
}

__global__ __launch_bounds__(128)
void lse_gemm(const bf16_t* __restrict__ A, const bf16_t* __restrict__ B,
              float* __restrict__ lse) {
  const int lane = threadIdx.x & 31;
  const int wave = threadIdx.x >> 5;
  const int rt   = blockIdx.x * 4 + wave;   // row tile (16 rows)
  const int h    = lane >> 4;               // lane half (0/1)
  const int ln   = lane & 15;

  // --- hoist the 8 A fragments (full 16x256 strip) into VGPRs -------------
  FragBF a[8];
  const bf16_t* ap = A + (size_t)(rt * 16 + ln) * FDIM + h * 8;
#pragma unroll
  for (int ks = 0; ks < 8; ++ks) {
    a[ks].f[0] = *(const v4f*)(ap + ks * 32);       // K = 32ks+8h   .. +7
    a[ks].f[1] = *(const v4f*)(ap + ks * 32 + 16);  // K = 32ks+16+8h.. +7
  }

  float rmax[8], rsum[8];
#pragma unroll
  for (int j = 0; j < 8; ++j) { rmax[j] = -__builtin_inff(); rsum[j] = 0.0f; }

  const bf16_t* bcol    = B + (size_t)ln * FDIM + h * 8;
  const size_t  tstride = (size_t)16 * FDIM;

  // --- double-buffered column-tile sweep (M/16 = 1250 tiles, even) --------
  // Tile ct+2 prologue load is unguarded: B has one 8KB pad tile at the end.
  FragBF bA[8], bB[8];
  load_btile(bA, bcol);
  for (int ct = 0; ct < M_ITEMS / 16; ct += 2) {
    load_btile(bB, bcol + (size_t)(ct + 1) * tstride);
    compute_tile(a, bA, rmax, rsum);
    load_btile(bA, bcol + (size_t)(ct + 2) * tstride);  // last iter: pad read
    compute_tile(a, bB, rmax, rsum);
  }

  // merge the 16 column-lanes of this half (rows identical within a half)
#pragma unroll
  for (int mask = 1; mask < 16; mask <<= 1) {
#pragma unroll
    for (int j = 0; j < 8; ++j) {
      float om = __shfl_xor(rmax[j], mask, 32);
      float os = __shfl_xor(rsum[j], mask, 32);
      float m  = fmaxf(rmax[j], om);
      rsum[j]  = rsum[j] * __expf(rmax[j] - m) + os * __expf(om - m);
      rmax[j]  = m;
    }
  }

  if (ln == 0) {
#pragma unroll
    for (int j = 0; j < 8; ++j)
      lse[rt * 16 + h * 8 + j] = rmax[j] + __logf(rsum[j]);
  }
}

// ---------------------------------------------------------------------------
// Gathered trace logits in exact fp32 (tiny: 8192*4 dot-256) + per-item loss
// ---------------------------------------------------------------------------
__global__ __launch_bounds__(256)
void sel_kernel(const int* __restrict__ labels, const float* __restrict__ feat,
                const float* __restrict__ mem, const int* __restrict__ tr,
                const float* __restrict__ lse, float* __restrict__ per_item) {
  const int lane = threadIdx.x & 31;
  const int wave = threadIdx.x >> 5;
  const int i    = blockIdx.x * 8 + wave;
  if (i >= N_ITEMS) return;

  const int lbl = labels[i];
  int t[TRACE_DEPTH];
#pragma unroll
  for (int d = 0; d < TRACE_DEPTH; ++d) t[d] = tr[lbl * TRACE_DEPTH + d];

  const float* f = feat + (size_t)i * FDIM;
  float acc[TRACE_DEPTH] = {0.f, 0.f, 0.f, 0.f};
  for (int c = lane; c < FDIM; c += 32) {
    float fv = f[c];
#pragma unroll
    for (int d = 0; d < TRACE_DEPTH; ++d)
      acc[d] += fv * mem[(size_t)t[d] * FDIM + c];
  }
#pragma unroll
  for (int mask = 1; mask < 32; mask <<= 1) {
#pragma unroll
    for (int d = 0; d < TRACE_DEPTH; ++d)
      acc[d] += __shfl_xor(acc[d], mask, 32);
  }
  if (lane == 0) {
    // sum_d d * sel_d, sel_d = dot/T ; depth weights 0,1,2,3, sum = 6
    float s   = (1.0f * acc[1] + 2.0f * acc[2] + 3.0f * acc[3]) * INV_T;
    float per = lse[i] - s * (1.0f / 6.0f);
    per_item[i] = (lbl != NUM_CLASSES) ? per : 0.0f;
  }
}

// ---------------------------------------------------------------------------
// Deterministic fixed-order final reduction (no float atomics)
// ---------------------------------------------------------------------------
__global__ __launch_bounds__(256)
void reduce_kernel(const float* __restrict__ per_item, float* __restrict__ out) {
  __shared__ float smem[256];
  float s = 0.0f;
  for (int i = threadIdx.x; i < N_ITEMS; i += 256) s += per_item[i];
  smem[threadIdx.x] = s;
  __syncthreads();
  for (int off = 128; off > 0; off >>= 1) {
    if (threadIdx.x < off) smem[threadIdx.x] += smem[threadIdx.x + off];
    __syncthreads();
  }
  if (threadIdx.x == 0) out[0] = smem[0] * LOSS_SCALE;
}

// ---------------------------------------------------------------------------
extern "C" void kernel_launch(void* const* d_in, const int* in_sizes, int n_in,
                              void* d_out, int out_size, void* d_ws,
                              size_t ws_size, hipStream_t stream) {
  const int*   gt   = (const int*)d_in[0];    // [N]
  const float* feat = (const float*)d_in[1];  // [N, F]
  const float* mem  = (const float*)d_in[2];  // [M, F]
  const int*   tidx = (const int*)d_in[3];    // [1000, 4]
  float*       out  = (float*)d_out;

  char* ws = (char*)d_ws;
  const size_t abytes   = (size_t)N_ITEMS * FDIM * 2;
  const size_t bbytes   = (size_t)M_ITEMS * FDIM * 2;
  const size_t padbytes = (size_t)16 * FDIM * 2;          // 1 pad tile (8KB)
  bf16_t* Abf = (bf16_t*)ws;
  bf16_t* Bbf = (bf16_t*)(ws + abytes);
  float*  lse = (float*)(ws + abytes + bbytes + padbytes);
  float*  per_item = lse + N_ITEMS;

  const int nA = N_ITEMS * FDIM;
  const int nB = M_ITEMS * FDIM;
  cvt_bf16_4<<<(nA / 4 + 255) / 256, 256, 0, stream>>>(feat, Abf, nA, INV_T);
  cvt_bf16_4<<<(nB / 4 + 255) / 256, 256, 0, stream>>>(mem,  Bbf, nB, 1.0f);

  lse_gemm<<<N_ITEMS / 16 / 4, 128, 0, stream>>>(Abf, Bbf, lse);

  sel_kernel<<<N_ITEMS / 8, 256, 0, stream>>>(gt, feat, mem, tidx, lse,
                                              per_item);

  reduce_kernel<<<1, 256, 0, stream>>>(per_item, out);
}
